// HybridSequenceStructureModel_14396730376433
// MI455X (gfx1250) — compile-verified, matrix operand order checked by reading, not run on hardware
//
#include <hip/hip_runtime.h>
#include <hip/hip_bf16.h>

#define NN    100000
#define NE    800000
#define NB    64
#define NF    9
#define HID   128
#define HEADS 4
#define HC    32
#define NL    4
#define SEQ   768
#define HD    256
#define ET    (NE + NN)          // edges + self loops
#define LN_EPS 1e-5f

typedef __attribute__((ext_vector_type(16))) __bf16          v16bf;
typedef __attribute__((ext_vector_type(16))) unsigned short  v16us;
typedef __attribute__((ext_vector_type(8)))  float           v8f;

__device__ __forceinline__ unsigned short f2bf(float f) {
  unsigned u = __float_as_uint(f);
  return (unsigned short)((u + 0x7FFFu + ((u >> 16) & 1u)) >> 16);  // RNE
}
// monotone float -> uint key for atomicMax over signed floats
__device__ __forceinline__ unsigned fkey(float f) {
  unsigned u = __float_as_uint(f);
  return (u & 0x80000000u) ? ~u : (u | 0x80000000u);
}
__device__ __forceinline__ float funkey(unsigned k) {
  return (k & 0x80000000u) ? __uint_as_float(k ^ 0x80000000u) : __uint_as_float(~k);
}

// ---------------------------------------------------------------- init
__global__ __launch_bounds__(256) void fill_f(float* p, float v, int n) {
  int g = blockIdx.x * 256 + threadIdx.x;
  if (g < n) p[g] = v;
}
__global__ __launch_bounds__(256) void layer_init(float* acc, float* ssum, unsigned* mkey) {
  int g = blockIdx.x * 256 + threadIdx.x;
  if (g < NN * HID) acc[g] = 0.f;
  if (g < NN * HEADS) { ssum[g] = 0.f; mkey[g] = fkey(-3.0e38f); }
}

// ---------------------------------------------------------------- node encoder
__global__ __launch_bounds__(256) void node_enc(const float* __restrict__ x,
                                                const float* __restrict__ W,
                                                const float* __restrict__ b,
                                                float* __restrict__ h) {
  int g = blockIdx.x * 256 + threadIdx.x;
  if (g >= NN * HID) return;
  int node = g >> 7, j = g & 127;
  const float* xr = x + node * NF;
  const float* wr = W + j * NF;
  float s = b[j];
#pragma unroll
  for (int k = 0; k < NF; ++k) s = fmaf(xr[k], wr[k], s);
  h[g] = fmaxf(s, 0.f);
}

// ---------------------------------------------------------------- xl = h @ W^T  (WMMA bf16)
// block = 256 threads = 8 waves; each block does a 128-row band, each wave 16 rows.
__global__ __launch_bounds__(256) void gemm_xl(const float* __restrict__ h,
                                               const float* __restrict__ Wl,   // [128][128] row-major
                                               float* __restrict__ xl) {
  __shared__ __align__(32) unsigned short sW[4 * 8 * 32 * 16];  // 32 KB, WMMA-B layout
  const int tid = threadIdx.x;

  // stage W into LDS pre-swizzled to the 16-bit B-operand lane layout
  for (int idx = tid; idx < 4 * 8 * 32 * 16; idx += 256) {
    int e    = idx & 15;
    int lane = (idx >> 4) & 31;
    int nT   = (idx >> 9) & 7;
    int kT   = idx >> 12;
    int j    = nT * 16 + (lane & 15);
    int khi  = (lane >> 4) * 8;
    int k    = kT * 32 + (e < 8 ? e + khi : e + 8 + khi);
    sW[idx]  = f2bf(Wl[j * HID + k]);
  }
  __syncthreads();

  const int wave  = tid >> 5;
  const int lane  = tid & 31;
  const int mBase = blockIdx.x * 128 + wave * 16;
  const int r     = mBase + (lane & 15);
  const int rr    = r < NN ? r : NN - 1;
  const int khi   = (lane >> 4) * 8;
  const bool full = (blockIdx.x + 1) * 128 <= NN;   // uniform: no row of this block OOB

  // load the 16x128 A band for this wave: 4 chunks of K=32
  v16us a[4];
#pragma unroll
  for (int c = 0; c < 4; ++c) {
    const float* base = h + (long)rr * HID + c * 32 + khi;
    float4 f0 = *(const float4*)(base);
    float4 f1 = *(const float4*)(base + 4);
    float4 f2 = *(const float4*)(base + 16);
    float4 f3 = *(const float4*)(base + 20);
    v16us t;
    t[0] = f2bf(f0.x);  t[1] = f2bf(f0.y);  t[2] = f2bf(f0.z);  t[3] = f2bf(f0.w);
    t[4] = f2bf(f1.x);  t[5] = f2bf(f1.y);  t[6] = f2bf(f1.z);  t[7] = f2bf(f1.w);
    t[8] = f2bf(f2.x);  t[9] = f2bf(f2.y);  t[10] = f2bf(f2.z); t[11] = f2bf(f2.w);
    t[12] = f2bf(f3.x); t[13] = f2bf(f3.y); t[14] = f2bf(f3.z); t[15] = f2bf(f3.w);
    a[c] = t;
  }

#pragma unroll
  for (int nT = 0; nT < 8; ++nT) {
    v8f acc = {};
#pragma unroll
    for (int c = 0; c < 4; ++c) {
      const v16us* bp = (const v16us*)&sW[(((c * 8 + nT) * 32 + lane) << 4)];
      v16bf av = __builtin_bit_cast(v16bf, a[c]);
      v16bf bv = __builtin_bit_cast(v16bf, *bp);
      acc = __builtin_amdgcn_wmma_f32_16x16x32_bf16(false, av, false, bv,
                                                    (short)0, acc, false, false);
    }
    const int col  = nT * 16 + (lane & 15);
    const int rowb = mBase + (lane >> 4) * 8;
    float* outp    = xl + (long)rowb * HID + col;
    if (full) {
      // hot path: one 64-bit base, 8 stores with immediate offsets, no exec masking
#pragma unroll
      for (int i = 0; i < 8; ++i) outp[i * HID] = acc[i];
    } else {
#pragma unroll
      for (int i = 0; i < 8; ++i)
        if (rowb + i < NN) outp[i * HID] = acc[i];
    }
  }
}

// ---------------------------------------------------------------- per-node attention scores
__global__ __launch_bounds__(256) void escore(const float* __restrict__ xl,
                                              const float* __restrict__ asrc,
                                              const float* __restrict__ adst,
                                              float* __restrict__ esrc,
                                              float* __restrict__ edst) {
  int g = blockIdx.x * 256 + threadIdx.x;
  if (g >= NN * HEADS) return;
  int node = g >> 2, hh = g & 3;
  const float* xr = xl + (long)node * HID + hh * HC;
  const float* as = asrc + hh * HC;
  const float* ad = adst + hh * HC;
  float s1 = 0.f, s2 = 0.f;
#pragma unroll
  for (int c = 0; c < HC; ++c) { s1 = fmaf(xr[c], as[c], s1); s2 = fmaf(xr[c], ad[c], s2); }
  esrc[g] = s1;
  edst[g] = s2;
}

// ---------------------------------------------------------------- edge: e = lrelu(es+ed), segment max
__global__ __launch_bounds__(256) void edge_e(const int* __restrict__ eidx,
                                              const float* __restrict__ esrc,
                                              const float* __restrict__ edst,
                                              float* __restrict__ ebuf,
                                              unsigned* __restrict__ mkey) {
  int g = blockIdx.x * 256 + threadIdx.x;
  if (g >= ET * HEADS) return;
  int eid = g >> 2, hh = g & 3;
  int s, d;
  if (eid < NE) { s = eidx[eid]; d = eidx[NE + eid]; } else { s = d = eid - NE; }
  float e = esrc[s * HEADS + hh] + edst[d * HEADS + hh];
  e = e > 0.f ? e : 0.2f * e;
  ebuf[g] = e;
  atomicMax(&mkey[d * HEADS + hh], fkey(e));
}

// ---------------------------------------------------------------- edge: p = exp(e-m), segment sum
__global__ __launch_bounds__(256) void edge_p(const int* __restrict__ eidx,
                                              float* __restrict__ ebuf,
                                              const unsigned* __restrict__ mkey,
                                              float* __restrict__ ssum) {
  int g = blockIdx.x * 256 + threadIdx.x;
  if (g >= ET * HEADS) return;
  int eid = g >> 2, hh = g & 3;
  int d = (eid < NE) ? eidx[NE + eid] : eid - NE;
  float p = expf(ebuf[g] - funkey(mkey[d * HEADS + hh]));
  ebuf[g] = p;
  atomicAdd(&ssum[d * HEADS + hh], p);
}

// ---------------------------------------------------------------- edge: acc[dst] += xl[src]*alpha (wave/edge)
__global__ __launch_bounds__(256) void edge_msg(const int* __restrict__ eidx,
                                                const float* __restrict__ ebuf,
                                                const float* __restrict__ ssum,
                                                const float* __restrict__ xl,
                                                float* __restrict__ acc) {
  int t = blockIdx.x * 256 + threadIdx.x;
  int eid = t >> 5, lane = t & 31;
  if (eid >= ET) return;
  int s, d;
  if (eid < NE) { s = eidx[eid]; d = eidx[NE + eid]; } else { s = d = eid - NE; }
  const float* xr = xl + (long)s * HID;
  __builtin_prefetch(xr + lane, 0, 1);
  float* ar = acc + (long)d * HID;
#pragma unroll
  for (int i = 0; i < HEADS; ++i) {   // HC == 32 == wave width: head i <-> iteration i
    float alpha = ebuf[eid * HEADS + i] / (ssum[d * HEADS + i] + 1e-16f);
    int c = i * HC + lane;
    atomicAdd(&ar[c], xr[c] * alpha);
  }
}

// ---------------------------------------------------------------- bias + layernorm + relu + residual (wave/node)
__global__ __launch_bounds__(256) void ln_res(const float* __restrict__ acc,
                                              const float* __restrict__ bias,
                                              const float* __restrict__ g_,
                                              const float* __restrict__ b_,
                                              float* __restrict__ h) {
  int t = blockIdx.x * 256 + threadIdx.x;
  int node = t >> 5, lane = t & 31;
  if (node >= NN) return;
  float v[4];
  float lsum = 0.f;
#pragma unroll
  for (int i = 0; i < 4; ++i) {
    int c = i * 32 + lane;
    v[i] = acc[(long)node * HID + c] + bias[c];
    lsum += v[i];
  }
#pragma unroll
  for (int off = 16; off >= 1; off >>= 1) lsum += __shfl_xor(lsum, off, 32);
  float mu = lsum * (1.f / HID);
  float d2 = 0.f;
#pragma unroll
  for (int i = 0; i < 4; ++i) { float dd = v[i] - mu; d2 += dd * dd; }
#pragma unroll
  for (int off = 16; off >= 1; off >>= 1) d2 += __shfl_xor(d2, off, 32);
  float rstd = rsqrtf(d2 * (1.f / HID) + LN_EPS);
#pragma unroll
  for (int i = 0; i < 4; ++i) {
    int c = i * 32 + lane;
    float hn = (v[i] - mu) * rstd * g_[c] + b_[c];
    h[(long)node * HID + c] += fmaxf(hn, 0.f);
  }
}

// ---------------------------------------------------------------- mean pool per graph
__global__ __launch_bounds__(256) void pool_sum(const float* __restrict__ h,
                                                const int* __restrict__ batch,
                                                float* __restrict__ pooled,
                                                float* __restrict__ cnt) {
  int g = blockIdx.x * 256 + threadIdx.x;
  if (g >= NN * HID) return;
  int node = g >> 7, c = g & 127;
  int b = batch[node];
  atomicAdd(&pooled[b * HID + c], h[g]);
  if (c == 0) atomicAdd(&cnt[b], 1.0f);
}
__global__ __launch_bounds__(256) void pool_div(float* __restrict__ pooled,
                                                const float* __restrict__ cnt) {
  int g = blockIdx.x * 256 + threadIdx.x;
  if (g >= NB * HID) return;
  pooled[g] /= fmaxf(cnt[g >> 7], 1.f);
}

// ---------------------------------------------------------------- generic small dense layer
__global__ __launch_bounds__(256) void lin(const float* __restrict__ in,
                                           const float* __restrict__ W,
                                           const float* __restrict__ bias,
                                           float* __restrict__ out,
                                           int K, int O, int relu) {
  int g = blockIdx.x * 256 + threadIdx.x;
  if (g >= NB * O) return;
  int b = g / O, o = g % O;
  const float* ir = in + b * K;
  const float* wr = W + o * K;
  float s = bias[o];
  for (int k = 0; k < K; ++k) s = fmaf(ir[k], wr[k], s);
  out[g] = relu ? fmaxf(s, 0.f) : s;
}

__global__ __launch_bounds__(256) void concat4(const float* a, const float* b,
                                               const float* c, const float* d,
                                               float* __restrict__ out) {
  int g = blockIdx.x * 256 + threadIdx.x;
  if (g >= NB * 4 * HD) return;
  int bb = g >> 10, i = g & 1023;
  float v;
  if (i < HD)            v = a[bb * HD + i];
  else if (i < 2 * HD)   v = b[bb * HD + i - HD];
  else if (i < 3 * HD)   v = c[bb * HD + i - 2 * HD];
  else                   v = d[bb * HD + i - 3 * HD];
  out[g] = v;
}

// ---------------------------------------------------------------- launcher
static inline int cdiv(long a, long b) { return (int)((a + b - 1) / b); }

extern "C" void kernel_launch(void* const* d_in, const int* in_sizes, int n_in,
                              void* d_out, int out_size, void* d_ws, size_t ws_size,
                              hipStream_t stream) {
  const float* seq_emb = (const float*)d_in[0];
  const float* x       = (const float*)d_in[1];
  const int*   eidx    = (const int*)  d_in[2];
  const int*   batch   = (const int*)  d_in[3];
  const float* ne_W    = (const float*)d_in[4];
  const float* ne_b    = (const float*)d_in[5];
  const float* gat_W   = (const float*)d_in[6];
  const float* gat_as  = (const float*)d_in[7];
  const float* gat_ad  = (const float*)d_in[8];
  const float* gat_b   = (const float*)d_in[9];
  const float* ln_g    = (const float*)d_in[10];
  const float* ln_b    = (const float*)d_in[11];
  const float* op1_W   = (const float*)d_in[12];
  const float* op1_b   = (const float*)d_in[13];
  const float* op2_W   = (const float*)d_in[14];
  const float* op2_b   = (const float*)d_in[15];
  const float* sp_W    = (const float*)d_in[16];
  const float* sp_b    = (const float*)d_in[17];
  const float* stp_W   = (const float*)d_in[18];
  const float* stp_b   = (const float*)d_in[19];
  const float* a1_Wi   = (const float*)d_in[20];
  const float* a1_bi   = (const float*)d_in[21];
  const float* a1_Wo   = (const float*)d_in[22];
  const float* a1_bo   = (const float*)d_in[23];
  const float* a2_Wi   = (const float*)d_in[24];
  const float* a2_bi   = (const float*)d_in[25];
  const float* a2_Wo   = (const float*)d_in[26];
  const float* a2_bo   = (const float*)d_in[27];
  const float* p1_W    = (const float*)d_in[28];
  const float* p1_b    = (const float*)d_in[29];
  const float* p2_W    = (const float*)d_in[30];
  const float* p2_b    = (const float*)d_in[31];
  const float* p3_W    = (const float*)d_in[32];
  const float* p3_b    = (const float*)d_in[33];

  // workspace layout
  float*    h      = (float*)d_ws;                  // NN*HID
  float*    xl     = h    + (long)NN * HID;         // NN*HID
  float*    acc    = xl   + (long)NN * HID;         // NN*HID
  float*    ebuf   = acc  + (long)NN * HID;         // ET*HEADS
  float*    ssum   = ebuf + (long)ET * HEADS;       // NN*HEADS
  unsigned* mkey   = (unsigned*)(ssum + (long)NN * HEADS);  // NN*HEADS
  float*    esrc   = (float*)(mkey + (long)NN * HEADS);     // NN*HEADS
  float*    edst   = esrc + (long)NN * HEADS;
  float*    pooled = edst + (long)NN * HEADS;       // NB*HID
  float*    cnt    = pooled + NB * HID;             // NB
  float*    tmp1   = cnt  + NB;                     // NB*HID
  float*    semb   = tmp1 + NB * HID;               // NB*HID
  float*    seqh   = semb + NB * HID;               // NB*HD
  float*    strh   = seqh + NB * HD;
  float*    v1     = strh + NB * HD;
  float*    v2     = v1   + NB * HD;
  float*    satt   = v2   + NB * HD;
  float*    tatt   = satt + NB * HD;
  float*    comb   = tatt + NB * HD;                // NB*4*HD
  float*    z1     = comb + NB * 4 * HD;            // NB*HD
  float*    z2     = z1   + NB * HD;                // NB*HD/2
  float*    out    = (float*)d_out;

  // node encoder
  node_enc<<<cdiv((long)NN * HID, 256), 256, 0, stream>>>(x, ne_W, ne_b, h);

  // GAT layers
  for (int l = 0; l < NL; ++l) {
    gemm_xl<<<cdiv(NN, 128), 256, 0, stream>>>(h, gat_W + (long)l * HID * HID, xl);
    escore<<<cdiv((long)NN * HEADS, 256), 256, 0, stream>>>(
        xl, gat_as + l * HEADS * HC, gat_ad + l * HEADS * HC, esrc, edst);
    layer_init<<<cdiv((long)NN * HID, 256), 256, 0, stream>>>(acc, ssum, mkey);
    edge_e<<<cdiv((long)ET * HEADS, 256), 256, 0, stream>>>(eidx, esrc, edst, ebuf, mkey);
    edge_p<<<cdiv((long)ET * HEADS, 256), 256, 0, stream>>>(eidx, ebuf, mkey, ssum);
    edge_msg<<<cdiv((long)ET * 32, 256), 256, 0, stream>>>(eidx, ebuf, ssum, xl, acc);
    ln_res<<<cdiv((long)NN * 32, 256), 256, 0, stream>>>(
        acc, gat_b + l * HID, ln_g + l * HID, ln_b + l * HID, h);
  }

  // mean pool
  fill_f<<<cdiv(NB * HID + NB, 256), 256, 0, stream>>>(pooled, 0.f, NB * HID + NB); // pooled+cnt contiguous
  pool_sum<<<cdiv((long)NN * HID, 256), 256, 0, stream>>>(h, batch, pooled, cnt);
  pool_div<<<cdiv(NB * HID, 256), 256, 0, stream>>>(pooled, cnt);

  // struct_emb = relu(pooled@op1^T+b)@op2^T+b
  lin<<<cdiv(NB * HID, 256), 256, 0, stream>>>(pooled, op1_W, op1_b, tmp1, HID, HID, 1);
  lin<<<cdiv(NB * HID, 256), 256, 0, stream>>>(tmp1, op2_W, op2_b, semb, HID, HID, 0);

  // projections
  lin<<<cdiv(NB * HD, 256), 256, 0, stream>>>(seq_emb, sp_W, sp_b, seqh, SEQ, HD, 0);
  lin<<<cdiv(NB * HD, 256), 256, 0, stream>>>(semb, stp_W, stp_b, strh, HID, HD, 0);

  // MHA with seq-len 1 on both sides: softmax(1 score)==1 -> out = (kv@Wv^T+bv)@Wo^T+bo
  lin<<<cdiv(NB * HD, 256), 256, 0, stream>>>(strh, a1_Wi + 2 * HD * HD, a1_bi + 2 * HD, v1, HD, HD, 0);
  lin<<<cdiv(NB * HD, 256), 256, 0, stream>>>(v1, a1_Wo, a1_bo, satt, HD, HD, 0);
  lin<<<cdiv(NB * HD, 256), 256, 0, stream>>>(seqh, a2_Wi + 2 * HD * HD, a2_bi + 2 * HD, v2, HD, HD, 0);
  lin<<<cdiv(NB * HD, 256), 256, 0, stream>>>(v2, a2_Wo, a2_bo, tatt, HD, HD, 0);

  // head
  concat4<<<cdiv(NB * 4 * HD, 256), 256, 0, stream>>>(satt, tatt, seqh, strh, comb);
  lin<<<cdiv(NB * HD, 256), 256, 0, stream>>>(comb, p1_W, p1_b, z1, 4 * HD, HD, 1);
  lin<<<cdiv(NB * (HD / 2), 256), 256, 0, stream>>>(z1, p2_W, p2_b, z2, HD, HD / 2, 1);
  lin<<<cdiv(NB, 256), 256, 0, stream>>>(z2, p3_W, p3_b, out, HD / 2, 1, 0);
}
